// AttentionBlock_21311627723062
// MI455X (gfx1250) — matrix operand order repaired and checked by reading, not compiled, not run on hardware
//
#include <hip/hip_runtime.h>
#include <hip/hip_bf16.h>
#include <math.h>

// AttentionBlock for MI455X (gfx1250).
// fp32 in/out; all GEMMs on V_WMMA_F32_16X16X32_BF16 via the bf16x3
// split-precision scheme (hi/lo planes, f32 accumulate) ~ fp32 accuracy at
// 16-bit matrix-pipe throughput. Operands are pre-split ONCE into bf16 hi/lo
// planes stored in fragment-native layouts so every WMMA fragment is two
// contiguous 16-byte loads (global_load_b128) -- no per-tile conversion VALU.
// B=16, C=256, N=H*W=1024, heads=4, dk=64.

typedef __bf16 v16bf  __attribute__((ext_vector_type(16)));
typedef __bf16 bf16x8 __attribute__((ext_vector_type(8)));
typedef float  v8f    __attribute__((ext_vector_type(8)));

#define CH     256
#define SEQ    1024
#define NHEADS 4
#define DK     64
#define QKVD   768   // 3 * NHEADS * DK
#define AD     256   // NHEADS * DK

union Frag {
  v16bf  v;
  bf16x8 h[2];
  unsigned int u[8];
};

__device__ __forceinline__ v8f wmma_bf16(v16bf a, v16bf b, v8f c) {
  // D(16x16,f32) = A(16x32,bf16) * B(32x16,bf16) + C(16x16,f32)
  return __builtin_amdgcn_wmma_f32_16x16x32_bf16(
      false, a, false, b, (short)0, c, false, false);
}

__device__ __forceinline__ v8f splat8(float v) {
  v8f r = {v, v, v, v, v, v, v, v};
  return r;
}

// Truncation split: hi = top 16 bits of a (exact), lo = bf16(a - hi).
__device__ __forceinline__ void split_store(float a, __bf16* hi, __bf16* lo) {
  union { unsigned short s; __bf16 b; } cv;
  const unsigned int ua = __float_as_uint(a);
  const float ah = __uint_as_float(ua & 0xffff0000u);
  cv.s = (unsigned short)(ua >> 16);
  *hi = cv.b;
  const float r = a - ah;
  cv.s = (unsigned short)(__float_as_uint(r) >> 16);
  *lo = cv.b;
}

// A-matrix (16x32, bf16) fragment: element e -> K = e + (e>=8)*8 + half*8.
// Contiguous runs: [half*8, +8) and [16 + half*8, +8).
__device__ __forceinline__ void loadA(Frag& f, const __bf16* row, int base, int half) {
  f.h[0] = *(const bf16x8*)(row + base + half * 8);
  f.h[1] = *(const bf16x8*)(row + base + 16 + half * 8);
}
// B-matrix (32x16, bf16) fragment: element e -> K = e + (e>=8)*8 + half*8 (same as A).
__device__ __forceinline__ void loadB(Frag& f, const __bf16* row, int base, int half) {
  f.h[0] = *(const bf16x8*)(row + base + half * 8);
  f.h[1] = *(const bf16x8*)(row + base + 16 + half * 8);
}

// max/sum over the 16-lane half-wave group (C-layout rows live per half)
__device__ __forceinline__ float hmax16(float v) {
  #pragma unroll
  for (int m = 1; m < 16; m <<= 1) v = fmaxf(v, __shfl_xor(v, m, 16));
  return v;
}
__device__ __forceinline__ float hsum16(float v) {
  #pragma unroll
  for (int m = 1; m < 16; m <<= 1) v += __shfl_xor(v, m, 16);
  return v;
}

// -------------------------------------------------------------------------
// Split kernel: x [B][C][N] f32 -> xT hi/lo bf16 planes [B][N][C].
// -------------------------------------------------------------------------
__global__ __launch_bounds__(256)
void split_x_kernel(const float* __restrict__ x,
                    __bf16* __restrict__ xhi, __bf16* __restrict__ xlo) {
  const int idx = blockIdx.x * 256 + threadIdx.x;   // b*C*N + c*N + n
  const int n = idx & (SEQ - 1);
  const int c = (idx >> 10) & (CH - 1);
  const int b = idx >> 18;
  const float a = x[idx];
  const size_t o = ((size_t)b * SEQ + n) * CH + c;
  split_store(a, xhi + o, xlo + o);
}

// -------------------------------------------------------------------------
// Split kernel: W [K][J] f32 -> WT hi/lo bf16 planes [J][K].
// -------------------------------------------------------------------------
__global__ __launch_bounds__(256)
void split_w_kernel(const float* __restrict__ W,
                    __bf16* __restrict__ whi, __bf16* __restrict__ wlo,
                    int K, int J) {
  const int idx = blockIdx.x * 256 + threadIdx.x;
  if (idx >= K * J) return;
  const int j = idx % J;
  const int k = idx / J;
  const size_t o = (size_t)j * K + k;
  split_store(W[idx], whi + o, wlo + o);
}

// -------------------------------------------------------------------------
// Kernel 1: qkv = xf @ W_qkv + b_qkv, written directly as bf16 hi/lo planes:
//   q,k: [b][h][n][d]   v: [b][h][d][n] (transposed for the PV B-fragment)
// One wave: 16x64 tile. grid = (768/64, 1024/16, B), block = 32.
// -------------------------------------------------------------------------
__global__ __launch_bounds__(32)
void qkv_proj_kernel(const __bf16* __restrict__ xhi, const __bf16* __restrict__ xlo,
                     const __bf16* __restrict__ whi, const __bf16* __restrict__ wlo,
                     const float* __restrict__ bqkv,
                     __bf16* __restrict__ qhi, __bf16* __restrict__ qlo,
                     __bf16* __restrict__ khi, __bf16* __restrict__ klo,
                     __bf16* __restrict__ vhi, __bf16* __restrict__ vlo) {
  const int lane = threadIdx.x;
  const int half = lane >> 4;
  const int l16  = lane & 15;
  const int j0   = blockIdx.x * 64;
  const int n0   = blockIdx.y * 16;
  const int b    = blockIdx.z;

  const __bf16* arow_h = xhi + ((size_t)b * SEQ + n0 + l16) * CH;
  const __bf16* arow_l = xlo + ((size_t)b * SEQ + n0 + l16) * CH;

  v8f acc[4];
  #pragma unroll
  for (int t = 0; t < 4; ++t) acc[t] = splat8(bqkv[j0 + t * 16 + l16]);

  for (int c0 = 0; c0 < CH; c0 += 32) {
    Frag ah, al;
    loadA(ah, arow_h, c0, half);
    loadA(al, arow_l, c0, half);
    #pragma unroll
    for (int t = 0; t < 4; ++t) {
      const __bf16* brow_h = whi + (size_t)(j0 + t * 16 + l16) * CH;
      const __bf16* brow_l = wlo + (size_t)(j0 + t * 16 + l16) * CH;
      Frag bh, bl;
      loadB(bh, brow_h, c0, half);
      loadB(bl, brow_l, c0, half);
      acc[t] = wmma_bf16(ah.v, bh.v, acc[t]);
      acc[t] = wmma_bf16(al.v, bh.v, acc[t]);
      acc[t] = wmma_bf16(ah.v, bl.v, acc[t]);
    }
  }

  // split-store into q/k/v hi-lo planes
  #pragma unroll
  for (int t = 0; t < 4; ++t) {
    const int jb     = j0 + t * 16;          // uniform per tile
    const int head   = jb / 192;
    const int rem    = jb % 192;
    const int region = rem / 64;             // 0:q 1:k 2:v
    const int d      = (rem % 64) + l16;
    const size_t bh_base = ((size_t)b * NHEADS + head);
    #pragma unroll
    for (int r = 0; r < 8; ++r) {
      const int row = n0 + r + half * 8;
      const float val = acc[t][r];
      if (region == 0) {
        const size_t o = (bh_base * SEQ + row) * DK + d;
        split_store(val, qhi + o, qlo + o);
      } else if (region == 1) {
        const size_t o = (bh_base * SEQ + row) * DK + d;
        split_store(val, khi + o, klo + o);
      } else {
        const size_t o = (bh_base * DK + d) * SEQ + row;   // transposed
        split_store(val, vhi + o, vlo + o);
      }
    }
  }
}

// -------------------------------------------------------------------------
// Kernel 2: flash attention per (batch, head, 16-query tile).
// 32 keys per block: two 16x16 score tiles, online softmax over 32 cols,
// P through LDS with u32 pair-packed hi/lo split, PV with K=32 bf16x3.
// Output split-stored into att hi/lo planes [b][n][256].
// grid = (1024/16, NHEADS, B), block = 32 (one wave).
// -------------------------------------------------------------------------
__global__ __launch_bounds__(32)
void attn_kernel(const __bf16* __restrict__ qhi, const __bf16* __restrict__ qlo,
                 const __bf16* __restrict__ khi, const __bf16* __restrict__ klo,
                 const __bf16* __restrict__ vhi, const __bf16* __restrict__ vlo,
                 __bf16* __restrict__ atthi, __bf16* __restrict__ attlo) {
  __shared__ float Plds[16][33];

  const int lane = threadIdx.x;
  const int half = lane >> 4;
  const int l16  = lane & 15;
  const int n0   = blockIdx.x * 16;
  const int h    = blockIdx.y;
  const int b    = blockIdx.z;
  const float scale = 0.125f;               // DK^-0.5

  const size_t bh  = (size_t)b * NHEADS + h;
  const __bf16* qrow_h = qhi + (bh * SEQ + n0 + l16) * DK;
  const __bf16* qrow_l = qlo + (bh * SEQ + n0 + l16) * DK;

  // Q tile: dk=64 -> two K=32 chunks held in registers.
  Frag aQh[2], aQl[2];
  #pragma unroll
  for (int c = 0; c < 2; ++c) {
    loadA(aQh[c], qrow_h, c * 32, half);
    loadA(aQl[c], qrow_l, c * 32, half);
  }

  v8f accO[4];
  #pragma unroll
  for (int t = 0; t < 4; ++t) accO[t] = splat8(0.0f);
  float rowm[8], rowl[8];
  #pragma unroll
  for (int r = 0; r < 8; ++r) { rowm[r] = -__builtin_inff(); rowl[r] = 0.0f; }

  for (int j0 = 0; j0 < SEQ; j0 += 32) {
    // ---- S = (Q @ K^T) * scale : two 16x16 key tiles ----
    v8f sv[2];
    #pragma unroll
    for (int kt = 0; kt < 2; ++kt) {
      const __bf16* krow_h = khi + (bh * SEQ + j0 + kt * 16 + l16) * DK;
      const __bf16* krow_l = klo + (bh * SEQ + j0 + kt * 16 + l16) * DK;
      v8f s = splat8(0.0f);
      #pragma unroll
      for (int c = 0; c < 2; ++c) {
        Frag bKh, bKl;                       // B[K=d][N=key] = Kmat[key][d]
        loadB(bKh, krow_h, c * 32, half);
        loadB(bKl, krow_l, c * 32, half);
        s = wmma_bf16(aQh[c].v, bKh.v, s);
        s = wmma_bf16(aQl[c].v, bKh.v, s);
        s = wmma_bf16(aQh[c].v, bKl.v, s);
      }
      sv[kt] = s;
    }

    // ---- online softmax over the 32 key columns ----
    float corr[8];
    v8f p0, p1;
    #pragma unroll
    for (int r = 0; r < 8; ++r) {
      const float s0 = sv[0][r] * scale;
      const float s1 = sv[1][r] * scale;
      const float tmax = fmaxf(hmax16(s0), hmax16(s1));
      const float mnew = fmaxf(rowm[r], tmax);
      corr[r] = __expf(rowm[r] - mnew);
      rowm[r] = mnew;
      const float e0 = __expf(s0 - mnew);
      const float e1 = __expf(s1 - mnew);
      p0[r] = e0;
      p1[r] = e1;
      rowl[r] = rowl[r] * corr[r] + hsum16(e0) + hsum16(e1);
    }

    // ---- P: C-layout -> LDS -> A-layout with u32 pair-packed hi/lo ----
    __syncthreads();
    #pragma unroll
    for (int r = 0; r < 8; ++r) {
      Plds[r + half * 8][l16]      = p0[r];
      Plds[r + half * 8][16 + l16] = p1[r];
    }
    __syncthreads();
    Frag phi, plo;
    #pragma unroll
    for (int i = 0; i < 8; ++i) {
      const int e0 = 2 * i;
      const int k0 = e0 + ((e0 >> 3) << 3) + half * 8;     // kmapA(e0)
      const float a0 = Plds[l16][k0];
      const float a1 = Plds[l16][k0 + 1];
      const unsigned int ua0 = __float_as_uint(a0);
      const unsigned int ua1 = __float_as_uint(a1);
      phi.u[i] = (ua1 & 0xffff0000u) | (ua0 >> 16);
      const float r0 = a0 - __uint_as_float(ua0 & 0xffff0000u);
      const float r1 = a1 - __uint_as_float(ua1 & 0xffff0000u);
      plo.u[i] = (__float_as_uint(r1) & 0xffff0000u) | (__float_as_uint(r0) >> 16);
    }

    // ---- O = O * corr + P @ V (K=32 chunk), V stored [d][n] ----
    #pragma unroll
    for (int t = 0; t < 4; ++t) {
      #pragma unroll
      for (int r = 0; r < 8; ++r) accO[t][r] *= corr[r];
      const __bf16* vrow_h = vhi + (bh * DK + t * 16 + l16) * SEQ;
      const __bf16* vrow_l = vlo + (bh * DK + t * 16 + l16) * SEQ;
      Frag vh, vl;                           // B[K=key][N=d], keys contiguous
      loadB(vh, vrow_h, j0, half);
      loadB(vl, vrow_l, j0, half);
      accO[t] = wmma_bf16(phi.v, vh.v, accO[t]);
      accO[t] = wmma_bf16(plo.v, vh.v, accO[t]);
      accO[t] = wmma_bf16(phi.v, vl.v, accO[t]);
    }
  }

  // ---- normalize and split-store att[b, n, h*64 + d] ----
  float inv[8];
  #pragma unroll
  for (int r = 0; r < 8; ++r) inv[r] = 1.0f / rowl[r];
  #pragma unroll
  for (int t = 0; t < 4; ++t) {
    #pragma unroll
    for (int r = 0; r < 8; ++r) {
      const int row = n0 + r + half * 8;
      const size_t o = ((size_t)b * SEQ + row) * AD + h * DK + t * 16 + l16;
      split_store(accO[t][r] * inv[r], atthi + o, attlo + o);
    }
  }
}

// -------------------------------------------------------------------------
// Kernel 3: out[b,c,n] = (att[b,n,:] @ W_out + b_out)[c] + x[b,c,n]
// grid = (256/64, 1024/16, B), block = 32.
// -------------------------------------------------------------------------
__global__ __launch_bounds__(32)
void out_proj_kernel(const __bf16* __restrict__ ahi, const __bf16* __restrict__ alo,
                     const __bf16* __restrict__ whi, const __bf16* __restrict__ wlo,
                     const float* __restrict__ bout,
                     const float* __restrict__ x,
                     float* __restrict__ out) {
  const int lane = threadIdx.x;
  const int half = lane >> 4;
  const int l16  = lane & 15;
  const int c0   = blockIdx.x * 64;
  const int n0   = blockIdx.y * 16;
  const int b    = blockIdx.z;

  const __bf16* arow_h = ahi + ((size_t)b * SEQ + n0 + l16) * AD;
  const __bf16* arow_l = alo + ((size_t)b * SEQ + n0 + l16) * AD;

  v8f acc[4];
  #pragma unroll
  for (int t = 0; t < 4; ++t) acc[t] = splat8(bout[c0 + t * 16 + l16]);

  for (int k0 = 0; k0 < AD; k0 += 32) {
    Frag ah, al;
    loadA(ah, arow_h, k0, half);
    loadA(al, arow_l, k0, half);
    #pragma unroll
    for (int t = 0; t < 4; ++t) {
      const __bf16* brow_h = whi + (size_t)(c0 + t * 16 + l16) * AD;
      const __bf16* brow_l = wlo + (size_t)(c0 + t * 16 + l16) * AD;
      Frag bh, bl;
      loadB(bh, brow_h, k0, half);
      loadB(bl, brow_l, k0, half);
      acc[t] = wmma_bf16(ah.v, bh.v, acc[t]);
      acc[t] = wmma_bf16(al.v, bh.v, acc[t]);
      acc[t] = wmma_bf16(ah.v, bl.v, acc[t]);
    }
  }

  const float* xb = x + (size_t)b * CH * SEQ;
  float* ob = out + (size_t)b * CH * SEQ;
  #pragma unroll
  for (int t = 0; t < 4; ++t) {
    #pragma unroll
    for (int r = 0; r < 8; ++r) {
      const int row = n0 + r + half * 8;
      const int c   = c0 + t * 16 + l16;
      ob[(size_t)c * SEQ + row] = acc[t][r] + xb[(size_t)c * SEQ + row];
    }
  }
}

// -------------------------------------------------------------------------
extern "C" void kernel_launch(void* const* d_in, const int* in_sizes, int n_in,
                              void* d_out, int out_size, void* d_ws, size_t ws_size,
                              hipStream_t stream) {
  const float* x    = (const float*)d_in[0];
  const float* Wqkv = (const float*)d_in[1];
  const float* bqkv = (const float*)d_in[2];
  const float* Wout = (const float*)d_in[3];
  const float* bout = (const float*)d_in[4];
  float* out = (float*)d_out;

  const int batch = in_sizes[0] / (CH * SEQ);   // 16

  // --- workspace: bf16 hi/lo planes (~85 MB total) ---
  __bf16* p = (__bf16*)d_ws;
  const size_t xtN   = (size_t)batch * SEQ * CH;
  const size_t wqN   = (size_t)QKVD * CH;
  const size_t woN   = (size_t)CH * AD;
  const size_t qkvN  = (size_t)batch * NHEADS * SEQ * DK;   // per plane
  const size_t attN  = (size_t)batch * SEQ * AD;

  __bf16* xThi = p;            p += xtN;
  __bf16* xTlo = p;            p += xtN;
  __bf16* wqhi = p;            p += wqN;
  __bf16* wqlo = p;            p += wqN;
  __bf16* wohi = p;            p += woN;
  __bf16* wolo = p;            p += woN;
  __bf16* qhi  = p;            p += qkvN;
  __bf16* qlo  = p;            p += qkvN;
  __bf16* khi  = p;            p += qkvN;
  __bf16* klo  = p;            p += qkvN;
  __bf16* vhi  = p;            p += qkvN;
  __bf16* vlo  = p;            p += qkvN;
  __bf16* athi = p;            p += attN;
  __bf16* atlo = p;            p += attN;

  // --- one-time operand splits ---
  split_x_kernel<<<(unsigned)(xtN / 256), 256, 0, stream>>>(x, xThi, xTlo);
  split_w_kernel<<<(unsigned)((wqN + 255) / 256), 256, 0, stream>>>(
      Wqkv, wqhi, wqlo, CH, QKVD);
  split_w_kernel<<<(unsigned)((woN + 255) / 256), 256, 0, stream>>>(
      Wout, wohi, wolo, AD, CH);

  // --- pipeline ---
  dim3 g1(QKVD / 64, SEQ / 16, batch);
  qkv_proj_kernel<<<g1, 32, 0, stream>>>(xThi, xTlo, wqhi, wqlo, bqkv,
                                         qhi, qlo, khi, klo, vhi, vlo);

  dim3 g2(SEQ / 16, NHEADS, batch);
  attn_kernel<<<g2, 32, 0, stream>>>(qhi, qlo, khi, klo, vhi, vlo, athi, atlo);

  dim3 g3(CH / 64, SEQ / 16, batch);
  out_proj_kernel<<<g3, 32, 0, stream>>>(athi, atlo, wohi, wolo, bout, x, out);
}
